// PlatonicLinear_13417477833309
// MI455X (gfx1250) — matrix operand order repaired and checked by reading, not compiled, not run on hardware
//
#include <hip/hip_runtime.h>

typedef __attribute__((ext_vector_type(16))) _Float16 v16h;
typedef __attribute__((ext_vector_type(8)))  float    v8f;

#define GORD 12
#define CH   16
#define KDIM 192
#define JDIM 192
#define LDS_ROW_DW 100            // 96 data dwords + 4 pad dwords per W row
#define LDS_ROW_B  (LDS_ROW_DW * 4)          // 400 B row stride
#define LDS_BYTES  (JDIM * LDS_ROW_B)        // 76800 B
#define ROWS_PER_WAVE 32          // two 16-row M tiles per wave
#define WAVES_PER_BLOCK 8
#define ITERS 2
#define ROWS_PER_BLOCK (ROWS_PER_WAVE * WAVES_PER_BLOCK * ITERS)  // 512

// ---------------------------------------------------------------------------
// Step 1: expand kernel(12,16,16) -> W(192,192) in f16 via the A4 group tables
//   W[a*16+oc, b*16+ic] = kernel[cayley[inv[b], a], oc, ic]
// ---------------------------------------------------------------------------
__global__ void build_w_f16(const float* __restrict__ kern,
                            const int*  __restrict__ cayley,
                            const int*  __restrict__ invidx,
                            _Float16*   __restrict__ Wh) {
    int idx = blockIdx.x * blockDim.x + threadIdx.x;
    if (idx >= JDIM * KDIM) return;
    int j  = idx / KDIM;
    int k  = idx - j * KDIM;
    int a  = j >> 4, oc = j & 15;
    int b  = k >> 4, ic = k & 15;
    int g  = cayley[invidx[b] * GORD + a];
    Wh[idx] = (_Float16)kern[(g * CH + oc) * CH + ic];
}

// Build an A fragment (16-bit A matrix 16x32 layout, ISA 7.12.2):
// lane holds row M = lane%16; halves 0..7 = K klo..klo+7, halves 8..15 = K klo+16..klo+23
__device__ __forceinline__ v16h load_a_frag(const float* __restrict__ rowp, int klo) {
    const float4* p = (const float4*)(rowp + klo);
    float4 p0 = p[0], p1 = p[1];
    const float4* q = (const float4*)(rowp + klo + 16);
    float4 q0 = q[0], q1 = q[1];
    v16h a;
    a[0]  = (_Float16)p0.x; a[1]  = (_Float16)p0.y; a[2]  = (_Float16)p0.z; a[3]  = (_Float16)p0.w;
    a[4]  = (_Float16)p1.x; a[5]  = (_Float16)p1.y; a[6]  = (_Float16)p1.z; a[7]  = (_Float16)p1.w;
    a[8]  = (_Float16)q0.x; a[9]  = (_Float16)q0.y; a[10] = (_Float16)q0.z; a[11] = (_Float16)q0.w;
    a[12] = (_Float16)q1.x; a[13] = (_Float16)q1.y; a[14] = (_Float16)q1.z; a[15] = (_Float16)q1.w;
    return a;
}

// ---------------------------------------------------------------------------
// Step 2: out[n, j] = sum_k x[n,k] * W[j,k] + bias[j%16]
// ---------------------------------------------------------------------------
__global__ __launch_bounds__(256) void platonic_gemm(
    const float*    __restrict__ x,
    const _Float16* __restrict__ Wh,
    const float*    __restrict__ bias,
    float*          __restrict__ out)
{
    extern __shared__ char smem_raw[];
    unsigned int* sW = (unsigned int*)smem_raw;

    // Stage W into LDS with padded row stride (bank-conflict-free B reads).
    const unsigned int* wdw = (const unsigned int*)Wh;
    for (int i = threadIdx.x; i < JDIM * (KDIM / 2); i += 256) {
        int r = i / (KDIM / 2);
        int c = i - r * (KDIM / 2);
        sW[r * LDS_ROW_DW + c] = wdw[i];
    }
    __syncthreads();

    const int lane  = threadIdx.x & 31;
    const int wave  = threadIdx.x >> 5;
    const int sel   = lane >> 4;    // which half-wave (K sub-range selector)
    const int lan16 = lane & 15;

    const float bval = bias[lan16]; // j % 16 == lane % 16 for every column tile

    // Loop-invariant per-lane LDS base for B fragments; all inner-loop DS
    // accesses become immediate offsets (jt*6400 + kb*64 <= 32320 < 64K).
    const char* bbase0 = smem_raw + lan16 * LDS_ROW_B + sel * 16;

    for (int t = 0; t < ITERS; ++t) {
        const int rowbase = blockIdx.x * ROWS_PER_BLOCK +
                            t * (ROWS_PER_WAVE * WAVES_PER_BLOCK) +
                            wave * ROWS_PER_WAVE;
        const float* xr0 = x + (size_t)(rowbase + lan16) * KDIM;
        const float* xr1 = xr0 + (size_t)16 * KDIM;

        // Hint the streaming read of the next row block into cache.
        __builtin_prefetch(xr0 + (size_t)ROWS_PER_WAVE * KDIM, 0, 0);

        for (int jh = 0; jh < 2; ++jh) {
            const char* bbase = bbase0 + jh * (6 * 16 * LDS_ROW_B);  // + jh*38400

            v8f acc0[6], acc1[6];
            #pragma unroll
            for (int jt = 0; jt < 6; ++jt)
                #pragma unroll
                for (int r = 0; r < 8; ++r) { acc0[jt][r] = bval; acc1[jt][r] = bval; }

            #pragma unroll
            for (int kb = 0; kb < 6; ++kb) {
                const int klo = kb * 32 + sel * 8;
                v16h a0 = load_a_frag(xr0, klo);
                v16h a1 = load_a_frag(xr1, klo);

                #pragma unroll
                for (int jt = 0; jt < 6; ++jt) {
                    // B fragment: K halves 8*sel..+7 and 16+8*sel..+7 of W row j (column of W^T)
                    const uint4* bp = (const uint4*)(bbase + jt * (16 * LDS_ROW_B) + kb * 64);
                    union { uint4 u[2]; v16h h; } bu;
                    bu.u[0] = bp[0];
                    bu.u[1] = bp[2];
                    v16h b = bu.h;

                    acc0[jt] = __builtin_amdgcn_wmma_f32_16x16x32_f16(
                        false, a0, false, b, (short)0, acc0[jt], false, false);
                    acc1[jt] = __builtin_amdgcn_wmma_f32_16x16x32_f16(
                        false, a1, false, b, (short)0, acc1[jt], false, false);
                }
            }

            // C/D layout: VGPR r -> row (r + 8*sel), col = lane%16
            #pragma unroll
            for (int jt = 0; jt < 6; ++jt) {
                const int j = (jh * 6 + jt) * 16 + lan16;
                float* o = out + (size_t)(rowbase + sel * 8) * JDIM + j;
                #pragma unroll
                for (int r = 0; r < 8; ++r) {
                    o[(size_t)r * JDIM]        = acc0[jt][r];
                    o[(size_t)(16 + r) * JDIM] = acc1[jt][r];
                }
            }
        }
    }
}

extern "C" void kernel_launch(void* const* d_in, const int* in_sizes, int n_in,
                              void* d_out, int out_size, void* d_ws, size_t ws_size,
                              hipStream_t stream) {
    const float* x      = (const float*)d_in[0];
    const float* kern   = (const float*)d_in[1];
    const float* bias   = (const float*)d_in[2];
    const int*   cayley = (const int*)d_in[3];
    const int*   invidx = (const int*)d_in[4];
    float*       out    = (float*)d_out;
    _Float16*    Wh     = (_Float16*)d_ws;   // 192*192*2 = 73728 B of scratch

    // Build expanded f16 weight matrix.
    build_w_f16<<<(JDIM * KDIM + 255) / 256, 256, 0, stream>>>(kern, cayley, invidx, Wh);

    const int N = in_sizes[0] / KDIM;            // 524288
    const int blocks = N / ROWS_PER_BLOCK;       // 1024
    platonic_gemm<<<blocks, 256, LDS_BYTES, stream>>>(x, Wh, bias, out);
}
